// MultiHeadAttention_17068200035070
// MI455X (gfx1250) — matrix operand order repaired and checked by reading, not compiled, not run on hardware
//
#include <hip/hip_runtime.h>

typedef __attribute__((ext_vector_type(16))) __bf16 v16bf;
typedef __attribute__((ext_vector_type(8)))  __bf16 v8bf;
typedef __attribute__((ext_vector_type(8)))  float  v8f;
typedef __attribute__((ext_vector_type(4))) unsigned int u32x4;
typedef __attribute__((ext_vector_type(8))) int i32x8;
typedef __attribute__((ext_vector_type(4))) int i32x4;

#define T_SEQ 2048
#define C_EMB 1024
#define NH    16
#define HD    64

static __device__ __forceinline__ __bf16 f2bf(float f) {
  union { float f; unsigned u; } x; x.f = f;
  unsigned r = x.u + 0x7FFFu + ((x.u >> 16) & 1u);
  unsigned short h = (unsigned short)(r >> 16);
  union { unsigned short u; __bf16 b; } y; y.u = h;
  return y.b;
}

static __device__ __forceinline__ v16bf cat16(v8bf lo, v8bf hi) {
  return __builtin_shufflevector(lo, hi, 0,1,2,3,4,5,6,7,8,9,10,11,12,13,14,15);
}

// ---------------- conversion kernels (one-time, memory-trivial) ----------------
__global__ void k_cvt(const float* __restrict__ src, __bf16* __restrict__ dst, int n) {
  int i = blockIdx.x * blockDim.x + threadIdx.x;
  if (i < n) dst[i] = f2bf(src[i]);
}

// dst[c][r] = src[r][c]  (store weights N-major so GEMM B-fragments are K-contiguous)
__global__ void k_cvt_T(const float* __restrict__ src, __bf16* __restrict__ dst,
                        int rows, int cols) {
  int i = blockIdx.x * blockDim.x + threadIdx.x;
  if (i < rows * cols) {
    int r = i / cols, c = i % cols;
    dst[c * rows + r] = f2bf(src[i]);
  }
}

// ---- Tensor Data Mover: DMA one 128-row x 32-col bf16 tile into LDS ----
// D# per CDNA5 ISA: data_size=2B (code 1), 2D tile,
// pad_enable, pad_interval=3 (every 16 DWORDs = one 64B row),
// pad_amount=3 (4 DWORDs = 16B)  ->  LDS row stride = 40 elements.
static __device__ __forceinline__ void tdm_tile_load(
    const __bf16* base, unsigned row0, unsigned k0,
    unsigned tensor_rows, unsigned lds_byte_off)
{
  unsigned long long ga = (unsigned long long)(uintptr_t)base
                        + ((unsigned long long)row0 * 1024ull + (unsigned long long)k0) * 2ull;
  u32x4 g0 = { 1u,                                   // count=1 (valid user descriptor)
               lds_byte_off,                         // lds_addr
               (unsigned)(ga & 0xFFFFFFFFu),         // global_addr[31:0]
               ((unsigned)((ga >> 32) & 0x1FFFFFFu)) // global_addr[56:32]
                 | 0x80000000u };                    // type=2 ("image")
  i32x8 g1 = { (int)((1u << 16)                      // data_size = 2 bytes
                   | (1u << 20)                      // pad_enable
                   | (3u << 22)                      // pad_interval: 16 DWORDs
                   | (3u << 25)),                    // pad_amount: 4 DWORDs
               (int)((1024u & 0xFFFFu) << 16),       // tensor_dim0[15:0]=1024
               (int)((tensor_rows & 0xFFFFu) << 16), // tensor_dim0[31:16]=0 | tensor_dim1[15:0]
               (int)(32u << 16)                      // tile_dim0 = 32
                 | (int)((tensor_rows >> 16) & 0xFFFFu), // tensor_dim1[31:16]
               (int)128,                             // tile_dim1 = 128, tile_dim2 = 0
               (int)1024,                            // tensor_dim0_stride[31:0] = 1024
               0, 0 };                               // stride hi / dim1_stride (2D: unused)
  i32x4 gz4 = { 0, 0, 0, 0 };
  i32x8 gz8 = { 0, 0, 0, 0, 0, 0, 0, 0 };
  __builtin_amdgcn_tensor_load_to_lds(g0, g1, gz4, gz4, gz8, 0);
}

// ---------------- WMMA GEMM: [M][1024] x [N][1024]^T, K = 1024 ----------------
// TDM-staged, double-buffered LDS. mode 0: QKV -> Q/K/V^T bf16; mode 1: fp32 out.
__global__ __launch_bounds__(256) void k_gemm(
    const __bf16* __restrict__ A,   // [M][1024] bf16, row-major
    const __bf16* __restrict__ BT,  // [N][1024] bf16 (pre-transposed weight)
    const float*  __restrict__ bias,
    int mode, int nrowsB,
    __bf16* __restrict__ Qb, __bf16* __restrict__ Kb, __bf16* __restrict__ Vt,
    float* __restrict__ out)
{
  // [buf][A|B][128*40] bf16: 4 tiles of 5120 elems = 40 KiB total
  __shared__ __bf16 smem[2 * 2 * 128 * 40];

  const int tid  = threadIdx.x;
  const int lane = tid & 31, half = lane >> 4, lr = lane & 15;
  const int wave = tid >> 5;
  const int wm = wave & 1, wn = wave >> 1;         // 2 x 4 wave grid
  const int m0 = blockIdx.y * 128;
  const int n0 = blockIdx.x * 128;

  const v8f zero = {0.f,0.f,0.f,0.f,0.f,0.f,0.f,0.f};
  v8f acc[4][2];
  #pragma unroll
  for (int i = 0; i < 4; ++i)
    #pragma unroll
    for (int j = 0; j < 2; ++j) acc[i][j] = zero;

  const unsigned ldsbase = (unsigned)(uintptr_t)(void*)smem;  // low 32 bits = LDS offset
  int cur = 0;

  // prologue: DMA tile 0 into buffer 0 (wave0 -> A tile, wave1 -> B tile)
  if (wave == 0) tdm_tile_load(A,  (unsigned)m0, 0u, 4096u, ldsbase);
  if (wave == 1) tdm_tile_load(BT, (unsigned)n0, 0u, (unsigned)nrowsB, ldsbase + 5120u * 2u);

  for (int k0 = 0; k0 < 1024; k0 += 32) {
    if (wave < 2) {
      if (k0 + 32 < 1024) {
        // stream next K tile into the other buffer, then wait for current tile
        unsigned boff = ldsbase + (unsigned)((cur ^ 1) * 10240) * 2u;
        if (wave == 0) tdm_tile_load(A,  (unsigned)m0, (unsigned)(k0 + 32), 4096u, boff);
        else           tdm_tile_load(BT, (unsigned)n0, (unsigned)(k0 + 32),
                                     (unsigned)nrowsB, boff + 5120u * 2u);
        __builtin_amdgcn_s_wait_tensorcnt(1);
      } else {
        __builtin_amdgcn_s_wait_tensorcnt(0);
      }
    }
    __syncthreads();

    const __bf16* sA = &smem[cur * 10240];
    const __bf16* sB = sA + 5120;

    v16bf af[4], bfr[2];
    #pragma unroll
    for (int i = 0; i < 4; ++i) {               // A frag: row = lane%16, K runs half*8 / 16+half*8
      const __bf16* pa = &sA[(wm * 64 + i * 16 + lr) * 40];
      af[i] = cat16(*(const v8bf*)&pa[half * 8], *(const v8bf*)&pa[16 + half * 8]);
    }
    #pragma unroll
    for (int j = 0; j < 2; ++j) {               // B frag: col = lane%16, K runs half*8 / 16+half*8
      const __bf16* pb = &sB[(wn * 32 + j * 16 + lr) * 40];
      bfr[j] = cat16(*(const v8bf*)&pb[half * 8], *(const v8bf*)&pb[16 + half * 8]);
    }
    #pragma unroll
    for (int i = 0; i < 4; ++i)
      #pragma unroll
      for (int j = 0; j < 2; ++j)
        acc[i][j] = __builtin_amdgcn_wmma_f32_16x16x32_bf16(
            false, af[i], false, bfr[j], (short)0, acc[i][j], false, false);
    __syncthreads();
    cur ^= 1;
  }

  // epilogue: C layout row = r + 8*half, col = lane%16
  #pragma unroll
  for (int i = 0; i < 4; ++i) {
    #pragma unroll
    for (int j = 0; j < 2; ++j) {
      const int gn = n0 + wn * 32 + j * 16 + lr;
      #pragma unroll
      for (int r = 0; r < 8; ++r) {
        const int gm = m0 + wm * 64 + i * 16 + half * 8 + r;
        float v = acc[i][j][r] + bias[gn];
        if (mode == 0) {
          int s = gn >> 10, cc = gn & 1023;
          int h = cc >> 6, d = cc & 63;
          int b = gm >> 11, t = gm & 2047;
          int bh = b * NH + h;
          __bf16 bv = f2bf(v);
          if (s == 0)      Qb[(bh * T_SEQ + t) * HD + d] = bv;
          else if (s == 1) Kb[(bh * T_SEQ + t) * HD + d] = bv;
          else             Vt[(bh * HD + d) * T_SEQ + t] = bv;   // V stored transposed
        } else {
          out[gm * 1024 + gn] = v;
        }
      }
    }
  }
}

// ---------------- flash attention: 1 wave per (b,h,16-query tile) ----------------
__global__ __launch_bounds__(32) void k_attn(
    const __bf16* __restrict__ Qb, const __bf16* __restrict__ Kb,
    const __bf16* __restrict__ Vt, __bf16* __restrict__ attn)
{
  __shared__ __bf16 sP[16 * 40];  // P tile C-layout -> A-layout relayout buffer

  const int lane = threadIdx.x & 31, half = lane >> 4, lr = lane & 15;
  const int bh = blockIdx.x >> 7;          // 0..31  (b*16 + h)
  const int q0 = (blockIdx.x & 127) << 4;  // query tile start
  const int b  = bh >> 4, h = bh & 15;

  // Q fragments held in registers for the whole kernel (M=16 queries, K = head dim)
  const __bf16* qp = Qb + (bh * T_SEQ + q0 + lr) * HD;
  const v16bf aq0 = cat16(*(const v8bf*)&qp[half * 8],      *(const v8bf*)&qp[16 + half * 8]);
  const v16bf aq1 = cat16(*(const v8bf*)&qp[32 + half * 8], *(const v8bf*)&qp[48 + half * 8]);

  const v8f zero = {0.f,0.f,0.f,0.f,0.f,0.f,0.f,0.f};
  v8f o[4];
  #pragma unroll
  for (int dt = 0; dt < 4; ++dt) o[dt] = zero;
  float mrow[8], lrow[8];
  #pragma unroll
  for (int r = 0; r < 8; ++r) { mrow[r] = -3.0e38f; lrow[r] = 0.f; }

  const int ntiles = (q0 >> 5) + 1;        // causal frontier, 32-key tiles
  for (int j = 0; j < ntiles; ++j) {
    const int kt = j << 5;

    // S = Q K^T for a 16x32 score tile (two 16x16 C frags, K-dim = 64 in 2 steps)
    v8f s[2];
    #pragma unroll
    for (int nt = 0; nt < 2; ++nt) {
      const __bf16* kp = Kb + (bh * T_SEQ + kt + nt * 16 + lr) * HD;
      v16bf b0 = cat16(*(const v8bf*)&kp[half * 8],      *(const v8bf*)&kp[16 + half * 8]);
      v16bf b1 = cat16(*(const v8bf*)&kp[32 + half * 8], *(const v8bf*)&kp[48 + half * 8]);
      v8f c = zero;
      c = __builtin_amdgcn_wmma_f32_16x16x32_bf16(false, aq0, false, b0, (short)0, c, false, false);
      c = __builtin_amdgcn_wmma_f32_16x16x32_bf16(false, aq1, false, b1, (short)0, c, false, false);
      s[nt] = c;
    }

    // online softmax; each row lives in one 16-lane group -> width-16 butterflies
    #pragma unroll
    for (int r = 0; r < 8; ++r) {
      const int qt = q0 + half * 8 + r;
      float e0 = s[0][r] * 0.125f;                    // 1/sqrt(64)
      float e1 = s[1][r] * 0.125f;
      if (kt + lr > qt)      e0 = -3.0e38f;           // causal mask
      if (kt + 16 + lr > qt) e1 = -3.0e38f;
      float mx = fmaxf(e0, e1);
      #pragma unroll
      for (int off = 1; off < 16; off <<= 1) mx = fmaxf(mx, __shfl_xor(mx, off, 16));
      float mn = fmaxf(mrow[r], mx);
      float al = __expf(mrow[r] - mn);
      mrow[r] = mn;
      e0 = __expf(e0 - mn);
      e1 = __expf(e1 - mn);
      float rs = e0 + e1;
      #pragma unroll
      for (int off = 1; off < 16; off <<= 1) rs += __shfl_xor(rs, off, 16);
      lrow[r] = lrow[r] * al + rs;
      #pragma unroll
      for (int dt = 0; dt < 4; ++dt) o[dt][r] *= al;
      sP[(half * 8 + r) * 40 + lr]      = f2bf(e0);   // relayout P via LDS
      sP[(half * 8 + r) * 40 + 16 + lr] = f2bf(e1);
    }
    asm volatile("s_wait_dscnt 0" ::: "memory");      // CDNA5 split DS counter

    // reload P as an A-fragment (16x32, K = key index)
    const __bf16* pp = &sP[lr * 40];
    v16bf pf = cat16(*(const v8bf*)&pp[half * 8], *(const v8bf*)&pp[16 + half * 8]);

    // O += P V  (B from pre-transposed V: rows are head-dim, contiguous in t)
    #pragma unroll
    for (int dt = 0; dt < 4; ++dt) {
      const __bf16* vp = Vt + (bh * HD + dt * 16 + lr) * T_SEQ + kt;
      v16bf bv = cat16(*(const v8bf*)&vp[half * 8], *(const v8bf*)&vp[16 + half * 8]);
      o[dt] = __builtin_amdgcn_wmma_f32_16x16x32_bf16(false, pf, false, bv, (short)0, o[dt], false, false);
    }
  }

  // normalize and store bf16 into [B][T][C] for the proj GEMM
  #pragma unroll
  for (int dt = 0; dt < 4; ++dt) {
    #pragma unroll
    for (int r = 0; r < 8; ++r) {
      const int t = q0 + half * 8 + r;
      float v = o[dt][r] / lrow[r];
      attn[(b * T_SEQ + t) * C_EMB + h * HD + dt * 16 + lr] = f2bf(v);
    }
  }
}

// ---------------- host orchestration ----------------
extern "C" void kernel_launch(void* const* d_in, const int* in_sizes, int n_in,
                              void* d_out, int out_size, void* d_ws, size_t ws_size,
                              hipStream_t stream) {
  const float* x      = (const float*)d_in[0];   // [2,2048,1024]
  const float* W_qkv  = (const float*)d_in[1];   // [1024,3072]
  const float* b_qkv  = (const float*)d_in[2];   // [3072]
  const float* W_proj = (const float*)d_in[3];   // [1024,1024]
  const float* b_proj = (const float*)d_in[4];   // [1024]
  float* out = (float*)d_out;                    // [2,2048,1024] fp32

  char* ws = (char*)d_ws;
  const size_t MiB = 1024 * 1024;
  __bf16* xb     = (__bf16*)(ws + 0 * MiB);   // 8 MiB  [4096][1024]
  __bf16* wqkvT  = (__bf16*)(ws + 8 * MiB);   // 6 MiB  [3072][1024]
  __bf16* wprojT = (__bf16*)(ws + 14 * MiB);  // 2 MiB  [1024][1024]
  __bf16* Qb     = (__bf16*)(ws + 16 * MiB);  // 8 MiB  [32][2048][64]
  __bf16* Kb     = (__bf16*)(ws + 24 * MiB);  // 8 MiB  [32][2048][64]
  __bf16* Vt     = (__bf16*)(ws + 32 * MiB);  // 8 MiB  [32][64][2048]
  __bf16* attn   = (__bf16*)(ws + 40 * MiB);  // 8 MiB  [4096][1024]

  k_cvt  <<<(4096 * 1024) / 256, 256, 0, stream>>>(x, xb, 4096 * 1024);
  k_cvt_T<<<(1024 * 3072) / 256, 256, 0, stream>>>(W_qkv,  wqkvT,  1024, 3072);
  k_cvt_T<<<(1024 * 1024) / 256, 256, 0, stream>>>(W_proj, wprojT, 1024, 1024);

  // QKV projection: M=4096, N=3072, K=1024
  k_gemm<<<dim3(24, 32), 256, 0, stream>>>(xb, wqkvT, b_qkv, 0, 3072, Qb, Kb, Vt, nullptr);

  // causal flash attention: 32 (b,h) x 128 query tiles
  k_attn<<<4096, 32, 0, stream>>>(Qb, Kb, Vt, attn);

  // output projection: M=4096, N=1024, K=1024 -> fp32 d_out
  k_gemm<<<dim3(8, 32), 256, 0, stream>>>(attn, wprojT, b_proj, 1, 1024,
                                          nullptr, nullptr, nullptr, out);
}
